// MyModel_79216376807648
// MI455X (gfx1250) — matrix operand order repaired and checked by reading, not compiled, not run on hardware
//
#include <hip/hip_runtime.h>
#include <math.h>

typedef __attribute__((ext_vector_type(16))) _Float16 v16h;
typedef __attribute__((ext_vector_type(8)))  _Float16 v8h;
typedef __attribute__((ext_vector_type(8)))  float    v8f;
typedef __attribute__((ext_vector_type(4)))  float    v4f;

#define B_ 32
#define N_ 512
#define E_ 128
#define H_ 64
#define NEGC 1e20f

__device__ __forceinline__ float sigmoidf_(float x) { return 1.0f / (1.0f + __expf(-x)); }

__device__ __forceinline__ v8f wmma_f16(v16h a, v16h b, v8f c) {
  return __builtin_amdgcn_wmma_f32_16x16x32_f16(false, a, false, b, (short)0, c, false, false);
}

// ---------------------------------------------------------------------------
// Pack W[K,N] (f32, row-major) into f16 WMMA B-fragment order:
//   Wp[(((ct*(K/32) + kblk)*32 + lane)*16 + i] = W[kblk*32 + (lane>>4)*8 + (i&7) + (i>>3)*16][ct*16 + (lane&15)]
// so each lane's fragment is one contiguous 32B v16h load.
// ---------------------------------------------------------------------------
__global__ void pack_w(const float* __restrict__ W, _Float16* __restrict__ Wp,
                       int N, int K, int total) {
  int t = blockIdx.x * 256 + threadIdx.x;
  if (t >= total) return;
  int i    = t & 15;
  int lane = (t >> 4) & 31;
  int blk  = t >> 9;                  // ct*(K/32) + kblk
  int kblk = blk % (K >> 5);
  int ct   = blk / (K >> 5);
  int k = kblk * 32 + (lane >> 4) * 8 + (i & 7) + (i >> 3) * 16;
  int n = ct * 16 + (lane & 15);
  Wp[t] = (_Float16)W[(size_t)k * N + n];
}

// ---------------------------------------------------------------------------
// Generic WMMA GEMM: out = act(A[M,K] @ W[K,N] + bias[N]); one 16x16 tile/wave.
// A f32 (converted to f16 in fragment loads), W pre-packed f16 fragments.
// OUTM: 0 = f32 store at row stride ldo; 2 = f16 packed-B store (for whg,
// hardcoded geometry M=B*512 rows -> batches, N=128 cols, K-dim 512).
// ---------------------------------------------------------------------------
template <int ACT, int OUTM, bool BIASED>
__global__ void gemm_wmma(const float* __restrict__ A, const _Float16* __restrict__ Wp,
                          const float* __restrict__ bias,
                          float* __restrict__ outF, _Float16* __restrict__ outH,
                          int M, int N, int K, int ldo, int tiles) {
  int wave = blockIdx.x * 8 + (threadIdx.x >> 5);
  if (wave >= tiles) return;                       // wave-uniform guard
  int tilesN = N >> 4;
  int tm = wave / tilesN, tn = wave % tilesN;
  int lane = threadIdx.x & 31;
  int l15 = lane & 15, kg = lane >> 4;
  int rowA = tm * 16 + l15;
  int col  = tn * 16 + l15;
  int kb   = K >> 5;
  v8f acc = {};
#pragma unroll 2
  for (int k0 = 0; k0 < K; k0 += 32) {
    const float* Ar = A + (size_t)rowA * K + k0 + kg * 8;
    v4f f0 = *(const v4f*)(Ar);
    v4f f1 = *(const v4f*)(Ar + 4);
    v4f f2 = *(const v4f*)(Ar + 16);
    v4f f3 = *(const v4f*)(Ar + 20);
    v16h a;
#pragma unroll
    for (int i = 0; i < 4; ++i) {
      a[i]      = (_Float16)f0[i];
      a[i + 4]  = (_Float16)f1[i];
      a[i + 8]  = (_Float16)f2[i];
      a[i + 12] = (_Float16)f3[i];
    }
    v16h b = *(const v16h*)(Wp + (((size_t)tn * kb + (k0 >> 5)) * 32 + lane) * 16);
    acc = wmma_f16(a, b, acc);
  }
  float bv = BIASED ? bias[col] : 0.0f;
#pragma unroll
  for (int v = 0; v < 8; ++v) {
    int r = tm * 16 + kg * 8 + v;
    float val = acc[v] + bv;
    if (ACT == 1) val = fmaxf(val, 0.0f);
    if (OUTM == 0) {
      outF[(size_t)r * ldo + col] = val;
    } else {
      // packed-B store: r = batch*512 + k-row, col in [0,128)
      int bb = r >> 9, kr = r & 511;
      int kblk = kr >> 5, wk = kr & 31;
      int laneP = ((wk >> 3) & 1) * 16 + (col & 15);
      int ii = (wk & 7) + (wk >> 4) * 8;
      size_t idx = ((((size_t)bb * 8 + (col >> 4)) * 16 + kblk) * 32 + laneP) * 16 + ii;
      outH[idx] = (_Float16)val;
    }
  }
}

// ---------------------------------------------------------------------------
// Batched attention GEMM with fused ELU epilogue (in-place on wh):
//   wh[b] = elu(wh[b] + attn[b][512,512] @ whg[b][512,128])
// attn f16 row-major; whg pre-packed f16 B fragments. Each wave computes a
// 16x32 tile (2 accumulators, A fragment reused). All fragment loads are
// contiguous b128s. 4096 waves total.
// ---------------------------------------------------------------------------
__global__ void attn_gemm_elu(const _Float16* __restrict__ attn,
                              const _Float16* __restrict__ whgT,
                              float* __restrict__ wh) {
  int wave = blockIdx.x * 8 + (threadIdx.x >> 5);
  int b = wave >> 7;           // 128 tiles per batch
  int tile = wave & 127;
  int tm = tile >> 2;          // 0..31 (M)
  int tn2 = tile & 3;          // 0..3  (N in 32-wide chunks)
  int lane = threadIdx.x & 31;
  int l15 = lane & 15, kg = lane >> 4;
  int row  = tm * 16 + l15;
  int col0 = tn2 * 32 + l15;
  int col1 = col0 + 16;
  int ct0 = tn2 * 2, ct1 = tn2 * 2 + 1;
  const _Float16* Ar = attn + ((size_t)b * N_ + row) * N_;
  const _Float16* Bt = whgT + (size_t)b * 65536;   // 8 ct * 16 kblk * 32 lane * 16
  v8f acc0 = {}, acc1 = {};
#pragma unroll 4
  for (int k0 = 0; k0 < N_; k0 += 32) {
    __builtin_prefetch(Ar + k0 + 128, 0, 0);      // global_prefetch_b8
    v8h alo = *(const v8h*)(Ar + k0 + kg * 8);
    v8h ahi = *(const v8h*)(Ar + k0 + 16 + kg * 8);
    v16h a = __builtin_shufflevector(alo, ahi, 0, 1, 2, 3, 4, 5, 6, 7,
                                     8, 9, 10, 11, 12, 13, 14, 15);
    int kblk = k0 >> 5;
    v16h b0 = *(const v16h*)(Bt + (((size_t)(ct0 * 16 + kblk)) * 32 + lane) * 16);
    v16h b1 = *(const v16h*)(Bt + (((size_t)(ct1 * 16 + kblk)) * 32 + lane) * 16);
    acc0 = wmma_f16(a, b0, acc0);
    acc1 = wmma_f16(a, b1, acc1);
  }
#pragma unroll
  for (int v = 0; v < 8; ++v) {
    int r = tm * 16 + kg * 8 + v;
    size_t base = ((size_t)b * N_ + r) * E_;
    float v0 = wh[base + col0] + acc0[v];
    float v1 = wh[base + col1] + acc1[v];
    wh[base + col0] = (v0 > 0.0f) ? v0 : (__expf(v0) - 1.0f);
    wh[base + col1] = (v1 > 0.0f) ? v1 : (__expf(v1) - 1.0f);
  }
}

// ---------------------------------------------------------------------------
// GRU recurrence (Keras reset_after=true). One block per batch element;
// rk [64,192] staged in LDS. 192 threads. Writes h_t into x[b, t, 64:128].
// ---------------------------------------------------------------------------
__global__ void gru_scan(const float* __restrict__ xp, const float* __restrict__ rk,
                         const float* __restrict__ gb, float* __restrict__ x) {
  __shared__ float rk_s[64 * 192];
  __shared__ float h_s[64];
  __shared__ float rec_s[192];
  int b = blockIdx.x, j = threadIdx.x;
  for (int idx = j; idx < 64 * 192; idx += 192) rk_s[idx] = rk[idx];
  if (j < 64) h_s[j] = 0.0f;
  float b1 = gb[192 + j];
  __syncthreads();
  for (int t = 0; t < N_; ++t) {
    float rec = b1;
#pragma unroll 8
    for (int k = 0; k < 64; ++k) rec += h_s[k] * rk_s[k * 192 + j];
    rec_s[j] = rec;
    __syncthreads();
    if (j < 64) {
      const float* xpt = xp + ((size_t)b * N_ + t) * 192;
      float z  = sigmoidf_(xpt[j] + rec_s[j]);
      float r  = sigmoidf_(xpt[64 + j] + rec_s[64 + j]);
      float hh = tanhf(xpt[128 + j] + r * rec_s[128 + j]);
      float hn = z * h_s[j] + (1.0f - z) * hh;
      h_s[j] = hn;
      x[((size_t)b * N_ + t) * E_ + H_ + j] = hn;
    }
    __syncthreads();
  }
}

__global__ void relu_copy(const float* __restrict__ in, float* __restrict__ out, int n) {
  int i = blockIdx.x * blockDim.x + threadIdx.x;
  if (i < n) out[i] = fmaxf(in[i], 0.0f);
}

__global__ void zero_k(float* __restrict__ p, int n) {
  int i = blockIdx.x * blockDim.x + threadIdx.x;
  if (i < n) p[i] = 0.0f;
}

// One wave per node row: ah1 = wh.a1_w + a1_b, ah2 = wh.a2_w + a2_b (wave32 shfl).
__global__ void ah_kernel(const float* __restrict__ wh, const float* __restrict__ a1w,
                          const float* __restrict__ a1b, const float* __restrict__ a2w,
                          const float* __restrict__ a2b, float* __restrict__ ah1,
                          float* __restrict__ ah2) {
  int row = blockIdx.x * 8 + (threadIdx.x >> 5);
  if (row >= B_ * N_) return;
  int lane = threadIdx.x & 31;
  const float* w = wh + (size_t)row * E_;
  float s1 = 0.0f, s2 = 0.0f;
  for (int e = lane; e < E_; e += 32) { float v = w[e]; s1 += v * a1w[e]; s2 += v * a2w[e]; }
#pragma unroll
  for (int m = 16; m >= 1; m >>= 1) { s1 += __shfl_xor(s1, m); s2 += __shfl_xor(s2, m); }
  if (lane == 0) { ah1[row] = s1 + a1b[0]; ah2[row] = s2 + a2b[0]; }
}

// One block per (b,i): leaky_relu(ah1_i + ah2_j) + mask, softmax over j, *A, f16 out.
// rev!=0 reads A transposed.
__global__ void attn_softmax(const float* __restrict__ cfg, const float* __restrict__ ah1,
                             const float* __restrict__ ah2, _Float16* __restrict__ attn,
                             int rev) {
  int i = blockIdx.x, b = blockIdx.y, tid = threadIdx.x;
  __shared__ float red[256];
  float e_i = ah1[b * N_ + i];
  int j0 = tid, j1 = tid + 256;
  float a0  = rev ? cfg[((size_t)b * N_ + j0) * N_ + i] : cfg[((size_t)b * N_ + i) * N_ + j0];
  float a1v = rev ? cfg[((size_t)b * N_ + j1) * N_ + i] : cfg[((size_t)b * N_ + i) * N_ + j1];
  float x0 = e_i + ah2[b * N_ + j0];
  float x1 = e_i + ah2[b * N_ + j1];
  x0 = (x0 >= 0.0f) ? x0 : 0.2f * x0;
  x1 = (x1 >= 0.0f) ? x1 : 0.2f * x1;
  x0 += a0  * NEGC - NEGC;                     // mask: 0 on edge, -1e20 off edge
  x1 += a1v * NEGC - NEGC;
  red[tid] = fmaxf(x0, x1); __syncthreads();
  for (int s = 128; s >= 1; s >>= 1) { if (tid < s) red[tid] = fmaxf(red[tid], red[tid + s]); __syncthreads(); }
  float mx = red[0]; __syncthreads();
  float p0 = __expf(x0 - mx), p1 = __expf(x1 - mx);
  red[tid] = p0 + p1; __syncthreads();
  for (int s = 128; s >= 1; s >>= 1) { if (tid < s) red[tid] += red[tid + s]; __syncthreads(); }
  float inv = 1.0f / red[0];
  _Float16* ar = attn + ((size_t)b * N_ + i) * N_;
  ar[j0] = (_Float16)(p0 * a0 * inv);
  ar[j1] = (_Float16)(p1 * a1v * inv);
}

// mid[b,e] += sum_i wh[b,i,e]
__global__ void node_reduce(const float* __restrict__ wh, float* __restrict__ mid) {
  int t = blockIdx.x * blockDim.x + threadIdx.x;
  if (t >= B_ * E_) return;
  int b = t >> 7, e = t & 127;
  float s = 0.0f;
  const float* p = wh + (size_t)b * N_ * E_ + e;
  for (int i = 0; i < N_; ++i) s += p[(size_t)i * E_];
  mid[t] += s;
}

// emb[b,o] = mid[b,:] @ W_out[:,o] + b_out[o]
__global__ void out_proj(const float* __restrict__ mid, const float* __restrict__ Wout,
                         const float* __restrict__ bout, float* __restrict__ emb) {
  int b = blockIdx.x, o = threadIdx.x;
  float s = bout[o];
  for (int e = 0; e < E_; ++e) s += mid[b * E_ + e] * Wout[e * 64 + o];
  emb[b * 64 + o] = s;
}

__global__ void final_cos(const float* __restrict__ emb, float* __restrict__ out) {
  int b = threadIdx.x;
  if (b >= B_) return;
  const float* e1 = emb + b * 64;
  const float* e2 = emb + B_ * 64 + b * 64;
  float d = 0.0f, n1 = 0.0f, n2 = 0.0f;
  for (int o = 0; o < 64; ++o) { d += e1[o] * e2[o]; n1 += e1[o] * e1[o]; n2 += e2[o] * e2[o]; }
  float c = d / (fmaxf(sqrtf(n1), 1e-12f) * fmaxf(sqrtf(n2), 1e-12f));
  out[b] = 0.5f * (1.0f + c);
}

extern "C" void kernel_launch(void* const* d_in, const int* in_sizes, int n_in,
                              void* d_out, int out_size, void* d_ws, size_t ws_size,
                              hipStream_t stream) {
  (void)in_sizes; (void)n_in; (void)out_size; (void)ws_size;
  const float* CFG[2] = {(const float*)d_in[0],  (const float*)d_in[4]};
  const float* LIT[2] = {(const float*)d_in[2],  (const float*)d_in[6]};
  const float* SEM[2] = {(const float*)d_in[3],  (const float*)d_in[7]};
  const float* WL[2]  = {(const float*)d_in[8],  (const float*)d_in[12]};
  const float* GK[2]  = {(const float*)d_in[9],  (const float*)d_in[13]};
  const float* GRK[2] = {(const float*)d_in[10], (const float*)d_in[14]};
  const float* GB[2]  = {(const float*)d_in[11], (const float*)d_in[15]};
  const float* a1w = (const float*)d_in[16]; const float* a1b = (const float*)d_in[17];
  const float* a2w = (const float*)d_in[18]; const float* a2b = (const float*)d_in[19];
  const float* Wg   = (const float*)d_in[20];
  const float* Wout = (const float*)d_in[21]; const float* bout = (const float*)d_in[22];

  // workspace layout (~48.3 MB)
  float* x   = (float*)d_ws;            // [B,N,E]        2,097,152 f32
  float* xp  = x  + 2097152;            // [B,N,192]      3,145,728 f32
  float* wh  = xp + 3145728;            // [B,N,E]        2,097,152 f32
  float* mid = wh + 2097152;            // [B,E]          4,096 f32
  float* emb = mid + 4096;              // [2,B,64]       4,096 f32
  float* ah1 = emb + 4096;              // [B,N]          16,384 f32
  float* ah2 = ah1 + 16384;             // [B,N]          16,384 f32
  _Float16* whgT = (_Float16*)(ah2 + 16384); // packed [B][8ct][16kblk][32lane][16]  2,097,152 f16
  _Float16* attn = whgT + 2097152;           // [B,N,N]   8,388,608 f16
  _Float16* WLp[2] = {attn + 8388608, attn + 8388608 + 4096};   // 64x64 packed each
  _Float16* GKp[2] = {WLp[1] + 4096, WLp[1] + 4096 + 12288};    // 64x192 packed each
  _Float16* Wgp    = GKp[1] + 12288;                            // 128x128 packed

  const int BN = B_ * N_;

  // Pack constant weights into WMMA B-fragment order (cheap, once per launch).
  for (int s = 0; s < 2; ++s) {
    pack_w<<<(64 * 64 + 255) / 256, 256, 0, stream>>>(WL[s], WLp[s], 64, 64, 64 * 64);
    pack_w<<<(64 * 192 + 255) / 256, 256, 0, stream>>>(GK[s], GKp[s], 192, 64, 64 * 192);
  }
  pack_w<<<(128 * 128 + 255) / 256, 256, 0, stream>>>(Wg, Wgp, 128, 128, 128 * 128);

  for (int s = 0; s < 2; ++s) {
    // x[:,:,0:64] = relu(LIT @ W_lit)   (ldo = E)
    { int tiles = (BN / 16) * (H_ / 16);
      gemm_wmma<1, 0, false><<<(tiles + 7) / 8, 256, 0, stream>>>(
          LIT[s], WLp[s], nullptr, x, nullptr, BN, H_, 64, E_, tiles); }
    // xp = SEM @ gru_k + b0
    { int tiles = (BN / 16) * (192 / 16);
      gemm_wmma<0, 0, true><<<(tiles + 7) / 8, 256, 0, stream>>>(
          SEM[s], GKp[s], GB[s], xp, nullptr, BN, 192, 64, 192, tiles); }
    // x[:,:,64:128] = GRU(xp)
    gru_scan<<<B_, 192, 0, stream>>>(xp, GRK[s], GB[s], x);

    zero_k<<<(B_ * E_ + 255) / 256, 256, 0, stream>>>(mid, B_ * E_);
    for (int dir = 0; dir < 2; ++dir) {
      relu_copy<<<(BN * E_ + 255) / 256, 256, 0, stream>>>(x, wh, BN * E_);
      for (int step = 0; step < 3; ++step) {
        ah_kernel<<<(BN + 7) / 8, 256, 0, stream>>>(wh, a1w, a1b, a2w, a2b, ah1, ah2);
        { int tiles = (BN / 16) * (E_ / 16);   // whgT = packB(f16(wh @ W_g))
          gemm_wmma<0, 2, false><<<(tiles + 7) / 8, 256, 0, stream>>>(
              wh, Wgp, nullptr, nullptr, whgT, BN, E_, E_, E_, tiles); }
        attn_softmax<<<dim3(N_, B_), 256, 0, stream>>>(CFG[s], ah1, ah2, attn, dir);
        attn_gemm_elu<<<512, 256, 0, stream>>>(attn, whgT, wh);
      }
      node_reduce<<<16, 256, 0, stream>>>(wh, mid);
    }
    out_proj<<<B_, 64, 0, stream>>>(mid, Wout, bout, emb + s * B_ * 64);
  }
  final_cos<<<1, 32, 0, stream>>>(emb, (float*)d_out);
}
